// baseGNN_2757369004523
// MI455X (gfx1250) — matrix operand order repaired and checked by reading, not compiled, not run on hardware
//
#include <hip/hip_runtime.h>

typedef __attribute__((ext_vector_type(16))) _Float16 v16h;
typedef __attribute__((ext_vector_type(8)))  float    v8f;

#define NEG_SLOPE 0.2f

__device__ __forceinline__ float leaky(float v) { return v > 0.f ? v : NEG_SLOPE * v; }

// ---------------------------------------------------------------------------
// Per-feature sum / sum-of-squares over rows (for featnorm). Wave reduction
// then one atomic per wave per feature.
// ---------------------------------------------------------------------------
template <int F>
__global__ void stats_kernel(const float* __restrict__ a, int n,
                             float* __restrict__ sum, float* __restrict__ sq) {
  float ls[F], lq[F];
#pragma unroll
  for (int f = 0; f < F; ++f) { ls[f] = 0.f; lq[f] = 0.f; }
  for (int i = blockIdx.x * blockDim.x + threadIdx.x; i < n;
       i += gridDim.x * blockDim.x) {
#pragma unroll
    for (int f = 0; f < F; ++f) {
      float v = a[(long)i * F + f];
      ls[f] += v;
      lq[f] += v * v;
    }
  }
#pragma unroll
  for (int f = 0; f < F; ++f) {
    for (int m = 16; m > 0; m >>= 1) {
      ls[f] += __shfl_xor(ls[f], m, 32);
      lq[f] += __shfl_xor(lq[f], m, 32);
    }
  }
  if ((threadIdx.x & 31) == 0) {
#pragma unroll
    for (int f = 0; f < F; ++f) {
      atomicAdd(&sum[f], ls[f]);
      atomicAdd(&sq[f], lq[f]);
    }
  }
}

// ws layout (floats): xsum@0 xsq@16 esum@32 esq@36 xmu@40 xinv@56 emu@72 einv@76
__global__ void finalize_stats(float* __restrict__ ws, float nN, float nE) {
  int t = threadIdx.x;
  if (t < 16) {
    float mu = ws[t] / nN;
    float var = ws[16 + t] / nN - mu * mu;
    ws[40 + t] = mu;
    ws[56 + t] = rsqrtf(var + 1e-5f);
  } else if (t < 20) {
    int f = t - 16;
    float mu = ws[32 + f] / nE;
    float var = ws[36 + f] / nE - mu * mu;
    ws[72 + f] = mu;
    ws[76 + f] = rsqrtf(var + 1e-5f);
  }
}

__global__ void count_kernel(const int* __restrict__ batch,
                             float* __restrict__ counts, int N) {
  int i = blockIdx.x * blockDim.x + threadIdx.x;
  if (i < N) atomicAdd(&counts[batch[i]], 1.f);
}

// ---------------------------------------------------------------------------
// Edge-message GEMM + scatter-add, WMMA f32_16x16x32_f16.
// One wave = one tile of 16 edges. msg[16,48] = concat(feat[src], e)[16,K] @ Wm[K,48].
// A layout (16-bit 16x32): lane l: row = l&15, element i -> K = kb + i + (i>=8)*8,
//   kb = 8*(l>>4). B layout (16-bit 32x16): col = l&15, element i -> K = kb + i + (i>=8)*8.
// C/D (f32 16x16): element i -> M = 8*(l>>4)+i, N = l&15.
// ---------------------------------------------------------------------------
template <int KIN, bool NORM>
__global__ void edge_wmma(const float* __restrict__ feat,   // [N,KIN] node feats
                          const float* __restrict__ eattr,  // [E,4]
                          const int* __restrict__ ei,       // [2,E] (src row, dst row)
                          const float* __restrict__ Wm,     // [KIN+4, 48] row-major
                          const float* __restrict__ ws,     // stats base
                          float* __restrict__ agg,          // [N,48]
                          int E) {
  constexpr int KTOT = KIN + 4;
  constexpr int NS = (KTOT <= 32) ? 1 : 2;  // k-steps of 32

  const int lane = threadIdx.x & 31;
  const int half = lane >> 4;
  const int m = lane & 15;
  const int kbA = half * 8;
  const int kbB = half * 16;

  const float* xmu = ws + 40;
  const float* xinv = ws + 56;
  const float* emu = ws + 72;
  const float* einv = ws + 76;

  // Preload B (Wm) fragments once per wave; reused across all edge tiles.
  v16h bf[3][NS];
#pragma unroll
  for (int nt = 0; nt < 3; ++nt)
#pragma unroll
    for (int s = 0; s < NS; ++s)
#pragma unroll
      for (int i = 0; i < 16; ++i) {
        int K = s * 32 + kbA + i + ((i >= 8) ? 8 : 0);
        float w = (K < KTOT) ? Wm[K * 48 + nt * 16 + m] : 0.f;
        bf[nt][s][i] = (_Float16)w;
      }

  const int* srcI = ei;
  const int* dstI = ei + E;
  const int ntiles = E >> 4;
  const int wid = blockIdx.x * (blockDim.x >> 5) + (threadIdx.x >> 5);
  const int nw = gridDim.x * (blockDim.x >> 5);

  for (int t = wid; t < ntiles; t += nw) {  // uniform per wave -> EXEC all ones
    const int e0 = t << 4;

    // Build A fragment: each lane gathers its K-slice of edge (e0 + m)'s features.
    const int eA = e0 + m;
    const int sA = srcI[eA];
    v16h af[NS];
#pragma unroll
    for (int s = 0; s < NS; ++s)
#pragma unroll
      for (int i = 0; i < 16; ++i) {
        int K = s * 32 + kbA + i + ((i >= 8) ? 8 : 0);
        float v = 0.f;
        if (K < KIN) {
          v = feat[(long)sA * KIN + K];
          if (NORM) v = (v - xmu[K]) * xinv[K];
        } else if (K < KTOT) {
          int f = K - KIN;
          v = (eattr[(long)eA * 4 + f] - emu[f]) * einv[f];
        }
        af[s][i] = (_Float16)v;
      }

    v8f acc0 = {}, acc1 = {}, acc2 = {};
#pragma unroll
    for (int s = 0; s < NS; ++s) {
      acc0 = __builtin_amdgcn_wmma_f32_16x16x32_f16(false, af[s], false, bf[0][s],
                                                    (short)0, acc0, false, false);
      acc1 = __builtin_amdgcn_wmma_f32_16x16x32_f16(false, af[s], false, bf[1][s],
                                                    (short)0, acc1, false, false);
      acc2 = __builtin_amdgcn_wmma_f32_16x16x32_f16(false, af[s], false, bf[2][s],
                                                    (short)0, acc2, false, false);
    }

    // Scatter-add msg rows into agg[dst]; all L2-resident on MI455X (192MB).
#pragma unroll
    for (int i = 0; i < 8; ++i) {
      const int M = half * 8 + i;
      const int d = dstI[e0 + M];
      float* base = agg + (long)d * 48;
      atomicAdd(base + 0 * 16 + m, acc0[i]);
      atomicAdd(base + 1 * 16 + m, acc1[i]);
      atomicAdd(base + 2 * 16 + m, acc2[i]);
    }
  }
}

// ---------------------------------------------------------------------------
// Node update: h[i][j] = leaky(x[i,:]@Wr[:,j] + agg[i][j] + b[j]).
// ---------------------------------------------------------------------------
__global__ void node_conv(const float* __restrict__ xin, const float* __restrict__ agg,
                          const float* __restrict__ Wr, const float* __restrict__ b,
                          const float* __restrict__ xmu, const float* __restrict__ xinv,
                          float* __restrict__ hout, int N, int KIN, int norm) {
  int t = blockIdx.x * blockDim.x + threadIdx.x;
  if (t >= N * 48) return;
  int i = t / 48, j = t % 48;
  float acc = agg[t] + b[j];
  for (int k = 0; k < KIN; ++k) {
    float v = xin[(long)i * KIN + k];
    if (norm) v = (v - xmu[k]) * xinv[k];
    acc += v * Wr[k * 48 + j];
  }
  hout[t] = leaky(acc);
}

__global__ void readout_kernel(const float* __restrict__ h,
                               const int* __restrict__ batch,
                               float* __restrict__ reads, int N, int l) {
  int t = blockIdx.x * blockDim.x + threadIdx.x;
  if (t >= N * 48) return;
  int i = t / 48, f = t % 48;
  atomicAdd(&reads[(long)batch[i] * 144 + l * 48 + f], h[t]);
}

// in [64,K] (optionally divided by counts), W [K,128], out [64,128] with leaky.
__global__ void mlp_dense(const float* __restrict__ in, const float* __restrict__ counts,
                          const float* __restrict__ W, const float* __restrict__ b,
                          float* __restrict__ out, int K) {
  int g = blockIdx.x, j = threadIdx.x;
  float inv = 1.f;
  if (counts) inv = 1.f / fmaxf(counts[g], 1.f);
  float acc = b[j];
  for (int k = 0; k < K; ++k) acc += (in[g * K + k] * inv) * W[k * 128 + j];
  out[g * 128 + j] = leaky(acc);
}

__global__ void mlp_out(const float* __restrict__ in, const float* __restrict__ Wf,
                        const float* __restrict__ bfv, float* __restrict__ out) {
  int t = threadIdx.x;  // 256 = 64 graphs * 4 classes
  int g = t >> 2, c = t & 3;
  float acc = bfv[c];
  for (int k = 0; k < 128; ++k) acc += in[g * 128 + k] * Wf[k * 4 + c];
  out[g * 4 + c] = acc;
}

// ---------------------------------------------------------------------------
extern "C" void kernel_launch(void* const* d_in, const int* in_sizes, int n_in,
                              void* d_out, int out_size, void* d_ws, size_t ws_size,
                              hipStream_t stream) {
  const float* x   = (const float*)d_in[0];
  const float* ea  = (const float*)d_in[1];
  const float* Wr0 = (const float*)d_in[2];
  const float* Wm0 = (const float*)d_in[3];
  const float* b0  = (const float*)d_in[4];
  const float* Wr1 = (const float*)d_in[5];
  const float* Wm1 = (const float*)d_in[6];
  const float* b1  = (const float*)d_in[7];
  const float* Wr2 = (const float*)d_in[8];
  const float* Wm2 = (const float*)d_in[9];
  const float* b2  = (const float*)d_in[10];
  const float* Wl1 = (const float*)d_in[11];
  const float* bl1 = (const float*)d_in[12];
  const float* Wl2 = (const float*)d_in[13];
  const float* bl2 = (const float*)d_in[14];
  const float* Wf  = (const float*)d_in[15];
  const float* bfv = (const float*)d_in[16];
  const int*   ei  = (const int*)d_in[17];
  const int*   bat = (const int*)d_in[18];

  const int N = in_sizes[0] / 16;
  const int E = in_sizes[1] / 4;

  float* wsf    = (float*)d_ws;
  float* counts = wsf + 128;
  float* reads  = wsf + 192;                 // 64*144
  float* mlp1   = reads + 64 * 144;          // 64*128
  float* mlp2   = mlp1 + 64 * 128;           // 64*128
  float* hA     = mlp2 + 64 * 128;           // N*48
  float* hB     = hA + (size_t)N * 48;       // N*48
  float* agg    = hB + (size_t)N * 48;       // N*48

  const int NT = N * 48;
  const int nodeBlocks = (NT + 255) / 256;
  const int edgeBlocks = 2048;  // 16384 waves grid-striding 100k edge tiles

  // Zero stats + counts + readout sums.
  hipMemsetAsync(wsf, 0, (size_t)(192 + 64 * 144) * sizeof(float), stream);

  stats_kernel<16><<<256, 256, 0, stream>>>(x, N, wsf + 0, wsf + 16);
  stats_kernel<4><<<512, 256, 0, stream>>>(ea, E, wsf + 32, wsf + 36);
  finalize_stats<<<1, 32, 0, stream>>>(wsf, (float)N, (float)E);
  count_kernel<<<(N + 255) / 256, 256, 0, stream>>>(bat, counts, N);

  // ---- layer 0 (normalized raw x as input, K=16) ----
  hipMemsetAsync(agg, 0, (size_t)NT * sizeof(float), stream);
  edge_wmma<16, true><<<edgeBlocks, 256, 0, stream>>>(x, ea, ei, Wm0, wsf, agg, E);
  node_conv<<<nodeBlocks, 256, 0, stream>>>(x, agg, Wr0, b0, wsf + 40, wsf + 56, hA, N, 16, 1);
  readout_kernel<<<nodeBlocks, 256, 0, stream>>>(hA, bat, reads, N, 0);

  // ---- layer 1 (hA -> hB, K=48) ----
  hipMemsetAsync(agg, 0, (size_t)NT * sizeof(float), stream);
  edge_wmma<48, false><<<edgeBlocks, 256, 0, stream>>>(hA, ea, ei, Wm1, wsf, agg, E);
  node_conv<<<nodeBlocks, 256, 0, stream>>>(hA, agg, Wr1, b1, wsf + 40, wsf + 56, hB, N, 48, 0);
  readout_kernel<<<nodeBlocks, 256, 0, stream>>>(hB, bat, reads, N, 1);

  // ---- layer 2 (hB -> hA, K=48) ----
  hipMemsetAsync(agg, 0, (size_t)NT * sizeof(float), stream);
  edge_wmma<48, false><<<edgeBlocks, 256, 0, stream>>>(hB, ea, ei, Wm2, wsf, agg, E);
  node_conv<<<nodeBlocks, 256, 0, stream>>>(hB, agg, Wr2, b2, wsf + 40, wsf + 56, hA, N, 48, 0);
  readout_kernel<<<nodeBlocks, 256, 0, stream>>>(hA, bat, reads, N, 2);

  // ---- per-graph MLP head ----
  mlp_dense<<<64, 128, 0, stream>>>(reads, counts, Wl1, bl1, mlp1, 144);
  mlp_dense<<<64, 128, 0, stream>>>(mlp1, nullptr, Wl2, bl2, mlp2, 128);
  mlp_out<<<1, 256, 0, stream>>>(mlp2, Wf, bfv, (float*)d_out);
}
